// SequenceClassifier_47682726920774
// MI455X (gfx1250) — matrix operand order repaired and checked by reading, not compiled, not run on hardware
//
#include <hip/hip_runtime.h>
#include <stdint.h>

#define TT 1024
#define BB 128
#define HH 64
#define EE 100
#define EP 128   // E padded to multiple of 32 (WMMA K)
#define G4 256   // 4*H
#define VV 50000

typedef __attribute__((ext_vector_type(16))) __bf16 v16bf;
typedef __attribute__((ext_vector_type(8)))  float  v8f;
typedef unsigned short u16;

union FragAB { v16bf v; uint4  q[2]; };
union FragC  { v8f   v; float4 q[2]; };

__device__ __forceinline__ u16 f2bf(float f) {
  unsigned u = __float_as_uint(f);
  u += 0x7FFFu + ((u >> 16) & 1u);          // round-to-nearest-even
  return (u16)(u >> 16);
}
__device__ __forceinline__ float sigmoidf_(float x) {
  return 1.0f / (1.0f + __expf(-x));
}

// A-fragment (16x32 bf16, row-major source, row already selected):
// lane r (&15) row; hi=lane>>4: K chunks [k0+hi*8..+7] -> v0..3, [k0+16+hi*8..+7] -> v4..7
__device__ __forceinline__ FragAB loadA(const u16* rowbase, int k0, int hi) {
  FragAB f;
  const u16* p = rowbase + k0 + hi * 8;
  f.q[0] = *(const uint4*)(p);
  f.q[1] = *(const uint4*)(p + 16);
  return f;
}
// B-fragment (32x16 bf16): lane n (&15) column of B == row of W; K chunks [k0+hi*8..+7], [k0+16+hi*8..+7]
__device__ __forceinline__ FragAB loadB(const u16* colbase, int k0, int hi) {
  FragAB f;
  const u16* p = colbase + k0 + hi * 8;
  f.q[0] = *(const uint4*)(p);
  f.q[1] = *(const uint4*)(p + 16);
  return f;
}
__device__ __forceinline__ v8f wmma_bf16(FragAB a, FragAB b, v8f c) {
  return __builtin_amdgcn_wmma_f32_16x16x32_bf16(false, a.v, false, b.v,
                                                 (short)0, c, false, false);
}

// ---------------- prep: fp32 -> bf16 weight conversion, K-padding, bias fold ----
__global__ void prep_weights(const float* __restrict__ Wih0, const float* __restrict__ Whh0,
                             const float* __restrict__ bih0, const float* __restrict__ bhh0,
                             const float* __restrict__ Wih1, const float* __restrict__ Whh1,
                             const float* __restrict__ bih1, const float* __restrict__ bhh1,
                             u16* __restrict__ Wih0B, u16* __restrict__ Whh0B,
                             u16* __restrict__ Wih1B, u16* __restrict__ Whh1B,
                             float* __restrict__ b0c, float* __restrict__ b1c) {
  int idx = blockIdx.x * blockDim.x + threadIdx.x;
  if (idx < G4 * EP) {                                   // Wih0: 256x100 -> 256x128 padded
    int row = idx >> 7, col = idx & 127;
    Wih0B[idx] = f2bf(col < EE ? Wih0[row * EE + col] : 0.0f);
  } else if (idx < G4 * EP + G4 * HH) {
    int j = idx - G4 * EP;
    Whh0B[j] = f2bf(Whh0[j]);
  } else if (idx < G4 * EP + 2 * G4 * HH) {
    int j = idx - (G4 * EP + G4 * HH);
    Wih1B[j] = f2bf(Wih1[j]);
  } else if (idx < G4 * EP + 3 * G4 * HH) {
    int j = idx - (G4 * EP + 2 * G4 * HH);
    Whh1B[j] = f2bf(Whh1[j]);
  } else if (idx < G4 * EP + 3 * G4 * HH + G4) {
    int g = idx - (G4 * EP + 3 * G4 * HH);
    b0c[g] = bih0[g] + bhh0[g];
  } else if (idx < G4 * EP + 3 * G4 * HH + 2 * G4) {
    int g = idx - (G4 * EP + 3 * G4 * HH + G4);
    b1c[g] = bih1[g] + bhh1[g];
  }
}

__global__ void prep_emb(const float* __restrict__ emb, u16* __restrict__ embB) {
  long long idx = (long long)blockIdx.x * blockDim.x + threadIdx.x;
  if (idx >= (long long)VV * EP) return;
  int row = (int)(idx >> 7), col = (int)(idx & 127);
  embB[idx] = f2bf(col < EE ? emb[row * EE + col] : 0.0f);
}

// ---------------- layer-0 input projection: XW0 = emb[x] @ Wih0^T + (bih0+bhh0) ----
// M ordering: row = bG*(T*16) + t*16 + bi  (so a 16-row tile = 16 batch rows at one t).
// Output stored as WMMA C-fragments: XW0[tile][lane][8] for direct reload in the scan.
__global__ __launch_bounds__(256) void proj0(
    const int* __restrict__ x, const u16* __restrict__ embB,
    const u16* __restrict__ WihB, const float* __restrict__ bc,
    float* __restrict__ XW0) {
  const int w = threadIdx.x >> 5, lane = threadIdx.x & 31;
  const int r = lane & 15, hi = lane >> 4;
  const int tileId = blockIdx.x * 8 + w;
  const int ntile = tileId & 15;
  const int t = (tileId >> 4) & (TT - 1);
  const int bG = tileId >> 14;

  const int tok = x[(bG * 16 + r) * TT + t];
  const u16* ap = embB + (size_t)tok * EP;
  const u16* bp = WihB + (size_t)(ntile * 16 + r) * EP;

  FragC acc;
  const float bias = bc[ntile * 16 + r];   // N = lane&15 for all C VGPRs
#pragma unroll
  for (int i = 0; i < 8; ++i) acc.v[i] = bias;

#pragma unroll
  for (int kk = 0; kk < EP; kk += 32) {
    FragAB a = loadA(ap, kk, hi);
    FragAB b = loadB(bp, kk, hi);
    acc.v = wmma_bf16(a, b, acc.v);
  }
  float* op = XW0 + (size_t)tileId * 256 + lane * 8;
  *(float4*)(op)     = acc.q[0];
  *(float4*)(op + 4) = acc.q[1];
}

// ---------------- fused two-layer recurrent scan -------------------------------
// 8 blocks (16 batch rows each) x 32 waves. Waves 0-15: layer0 (step t),
// waves 16-31: layer1 (step t-1, K=128 concat [h0,h1] @ [Wih1;Whh1]^T).
// h/c live in LDS (h stride padded to 72 bf16 -> conflict-free ds_load_b128).
__global__ __launch_bounds__(1024) void lstm_fused_scan(
    const float* __restrict__ XW0, const u16* __restrict__ Whh0B,
    const u16* __restrict__ Wih1B, const u16* __restrict__ Whh1B,
    const float* __restrict__ b1c, float* __restrict__ lastH) {
  __shared__ u16   h0_lds[16 * 72];
  __shared__ u16   h1_lds[16 * 72];
  __shared__ float c0_lds[16 * 64];
  __shared__ float c1_lds[16 * 64];
  __shared__ float g0_lds[16 * 32 * 8];
  __shared__ float g1_lds[16 * 32 * 8];

  const int tid  = threadIdx.x;
  const int w    = tid >> 5;
  const int lane = tid & 31;
  const int r    = lane & 15;
  const int hi   = lane >> 4;
  const int layer = w >> 4;
  const int nt    = w & 15;
  const int bG    = blockIdx.x;

  // Weight B-fragments resident in VGPRs for the whole scan.
  FragAB Bw0, Bw1, Bw2, Bw3;
  float bias1 = 0.0f;
  if (layer == 0) {
    const u16* bp = Whh0B + (size_t)(nt * 16 + r) * HH;
    Bw0 = loadB(bp, 0, hi);
    Bw1 = loadB(bp, 32, hi);
    Bw2 = Bw0; Bw3 = Bw1;                       // unused for layer 0
  } else {
    const u16* bpi = Wih1B + (size_t)(nt * 16 + r) * HH;
    const u16* bph = Whh1B + (size_t)(nt * 16 + r) * HH;
    Bw0 = loadB(bpi, 0, hi);
    Bw1 = loadB(bpi, 32, hi);
    Bw2 = loadB(bph, 0, hi);
    Bw3 = loadB(bph, 32, hi);
    bias1 = b1c[nt * 16 + r];
  }

  for (int i = tid; i < 16 * 72; i += 1024) { h0_lds[i] = 0; h1_lds[i] = 0; }
  for (int i = tid; i < 16 * 64; i += 1024) { c0_lds[i] = 0.0f; c1_lds[i] = 0.0f; }
  __syncthreads();

  for (int iter = 0; iter <= TT; ++iter) {
    const bool run0 = (layer == 0) && (iter < TT);
    const bool run1 = (layer == 1) && (iter >= 1);

    // ---- phase A: GEMMs (read pre-update h, write gate fragments) ----
    if (run0) {
      const float* xwp = XW0 + (((size_t)bG * TT + iter) * 16 + nt) * 256 + lane * 8;
      FragC acc;
      acc.q[0] = *(const float4*)(xwp);
      acc.q[1] = *(const float4*)(xwp + 4);
      const u16* hp = h0_lds + r * 72;
      FragAB A0 = loadA(hp, 0, hi);
      FragAB A1 = loadA(hp, 32, hi);
      acc.v = wmma_bf16(A0, Bw0, acc.v);
      acc.v = wmma_bf16(A1, Bw1, acc.v);
      float* gp = &g0_lds[(nt * 32 + lane) * 8];
      *(float4*)(gp)     = acc.q[0];
      *(float4*)(gp + 4) = acc.q[1];
    }
    if (run1) {
      FragC acc;
#pragma unroll
      for (int i = 0; i < 8; ++i) acc.v[i] = bias1;
      const u16* hp0 = h0_lds + r * 72;   // h0_{iter-1}
      const u16* hp1 = h1_lds + r * 72;   // h1_{iter-2}
      FragAB A0 = loadA(hp0, 0, hi);
      FragAB A1 = loadA(hp0, 32, hi);
      FragAB A2 = loadA(hp1, 0, hi);
      FragAB A3 = loadA(hp1, 32, hi);
      acc.v = wmma_bf16(A0, Bw0, acc.v);
      acc.v = wmma_bf16(A1, Bw1, acc.v);
      acc.v = wmma_bf16(A2, Bw2, acc.v);
      acc.v = wmma_bf16(A3, Bw3, acc.v);
      float* gp = &g1_lds[(nt * 32 + lane) * 8];
      *(float4*)(gp)     = acc.q[0];
      *(float4*)(gp + 4) = acc.q[1];
    }
    __syncthreads();

    // ---- phase B: elementwise LSTM cell update (write h/c for next step) ----
    if (run0) {
#pragma unroll
      for (int e = 0; e < 2; ++e) {
        const int cell = tid * 2 + e;            // tid in [0,512)
        const int bi = cell >> 6, n = cell & 63;
        float gv[4];
#pragma unroll
        for (int q = 0; q < 4; ++q) {            // i,f,g,o at cols q*64+n
          const int col = q * 64 + n;
          const int ww = col >> 4;
          const int l  = (col & 15) + ((bi >> 3) << 4);
          gv[q] = g0_lds[(ww * 32 + l) * 8 + (bi & 7)];
        }
        const float c = sigmoidf_(gv[1]) * c0_lds[cell] + sigmoidf_(gv[0]) * tanhf(gv[2]);
        c0_lds[cell] = c;
        h0_lds[bi * 72 + n] = f2bf(sigmoidf_(gv[3]) * tanhf(c));
      }
    }
    if (run1) {
#pragma unroll
      for (int e = 0; e < 2; ++e) {
        const int cell = (tid - 512) * 2 + e;
        const int bi = cell >> 6, n = cell & 63;
        float gv[4];
#pragma unroll
        for (int q = 0; q < 4; ++q) {
          const int col = q * 64 + n;
          const int ww = col >> 4;
          const int l  = (col & 15) + ((bi >> 3) << 4);
          gv[q] = g1_lds[(ww * 32 + l) * 8 + (bi & 7)];
        }
        const float c = sigmoidf_(gv[1]) * c1_lds[cell] + sigmoidf_(gv[0]) * tanhf(gv[2]);
        c1_lds[cell] = c;
        const float h = sigmoidf_(gv[3]) * tanhf(c);
        h1_lds[bi * 72 + n] = f2bf(h);
        if (iter == TT)                           // h1_{T-1}: the classifier input
          lastH[((size_t)bG * 16 + bi) * HH + n] = h;
      }
    }
    __syncthreads();
  }
}

// ---------------- head: sigmoid(relu(last @ W1^T + b1) @ W2^T + b2) -------------
__global__ __launch_bounds__(128) void head(
    const float* __restrict__ lastH, const float* __restrict__ W1,
    const float* __restrict__ b1, const float* __restrict__ W2,
    const float* __restrict__ b2, float* __restrict__ out) {
  const int b = threadIdx.x;
  float last[HH];
#pragma unroll
  for (int k = 0; k < HH; ++k) last[k] = lastH[(size_t)b * HH + k];
  float acc2 = b2[0];
  for (int j = 0; j < HH; ++j) {
    float s = b1[j];
#pragma unroll
    for (int k = 0; k < HH; ++k) s += last[k] * W1[j * HH + k];
    acc2 += fmaxf(s, 0.0f) * W2[j];
  }
  out[b] = sigmoidf_(acc2);
}

// ---------------- launch --------------------------------------------------------
extern "C" void kernel_launch(void* const* d_in, const int* in_sizes, int n_in,
                              void* d_out, int out_size, void* d_ws, size_t ws_size,
                              hipStream_t stream) {
  const int*   x    = (const int*)  d_in[0];
  const float* emb  = (const float*)d_in[1];
  const float* Wih0 = (const float*)d_in[2];
  const float* Whh0 = (const float*)d_in[3];
  const float* bih0 = (const float*)d_in[4];
  const float* bhh0 = (const float*)d_in[5];
  const float* Wih1 = (const float*)d_in[6];
  const float* Whh1 = (const float*)d_in[7];
  const float* bih1 = (const float*)d_in[8];
  const float* bhh1 = (const float*)d_in[9];
  const float* W1   = (const float*)d_in[10];
  const float* b1   = (const float*)d_in[11];
  const float* W2   = (const float*)d_in[12];
  const float* b2   = (const float*)d_in[13];
  float* out = (float*)d_out;

  char* ws = (char*)d_ws;
  size_t off = 0;
  auto carve = [&](size_t bytes) -> void* {
    void* p = ws + off;
    off = (off + bytes + 255) & ~(size_t)255;
    return p;
  };
  u16*   embB  = (u16*)  carve((size_t)VV * EP * 2);     // 12.8 MB
  u16*   Wih0B = (u16*)  carve((size_t)G4 * EP * 2);
  u16*   Whh0B = (u16*)  carve((size_t)G4 * HH * 2);
  u16*   Wih1B = (u16*)  carve((size_t)G4 * HH * 2);
  u16*   Whh1B = (u16*)  carve((size_t)G4 * HH * 2);
  float* b0c   = (float*)carve(G4 * 4);
  float* b1c   = (float*)carve(G4 * 4);
  float* XW0   = (float*)carve((size_t)BB * TT * G4 * 4); // 134 MB (< 192 MB L2)
  float* lastH = (float*)carve((size_t)BB * HH * 4);
  (void)ws_size; (void)in_sizes; (void)n_in; (void)out_size;

  prep_weights<<<322, 256, 0, stream>>>(Wih0, Whh0, bih0, bhh0, Wih1, Whh1,
                                        bih1, bhh1, Wih0B, Whh0B, Wih1B, Whh1B,
                                        b0c, b1c);
  prep_emb<<<(VV * EP) / 256, 256, 0, stream>>>(emb, embB);
  proj0<<<(BB / 16) * TT * 16 / 8, 256, 0, stream>>>(x, embB, Wih0B, b0c, XW0);
  lstm_fused_scan<<<BB / 16, 1024, 0, stream>>>(XW0, Whh0B, Wih1B, Whh1B, b1c, lastH);
  head<<<1, 128, 0, stream>>>(lastH, W1, b1, W2, b2, out);
}
